// RENL2_52802327937741
// MI455X (gfx1250) — matrix operand order repaired and checked by reading, not compiled, not run on hardware
//
#include <hip/hip_runtime.h>
#include <hip/hip_bf16.h>

typedef __attribute__((ext_vector_type(16))) _Float16 v16h;
typedef __attribute__((ext_vector_type(8)))  _Float16 v8h;
typedef __attribute__((ext_vector_type(8)))  float    v8f;
typedef __attribute__((ext_vector_type(4)))  int      v4i;

#define NTOT 262144
#define ROWS_PER_WG 128

// Pre-converted weight blob layout (identical in d_ws and in LDS)
#define OFF_C1    0        // 64x64 f16          (8192 B)
#define OFF_A     8192     // 64x64 f16          (8192 B)
#define OFF_B1    16384    // 64x64 f16          (8192 B)
#define OFF_D12   24576    // 64x32 f16 K-padded (4096 B)
#define OFF_B2    28672    // 64x32 f16 K-padded (4096 B)
#define OFF_D11M  32768    // 4 x 16x64 f16 block-masked D11 (8192 B)
#define OFF_DDIAG 40960    // 4 x 16x16 f32 transposed masked diag (4096 B)
#define OFF_BV    45056    // 64 f32 (256 B)
#define OFF_BX    45312    // 64 f32 (256 B)
#define BLOB_BYTES 45568

#if defined(__has_builtin)
#if __has_builtin(__builtin_amdgcn_global_load_async_to_lds_b128)
#define HAVE_ASYNC_LDS 1
#endif
#endif

// D = A(16x32 f16) * B(32x16 f16) + C(16x16 f32)
__device__ __forceinline__ v8f wmma_f16(v16h a, v16h b, v8f c) {
  return __builtin_amdgcn_wmma_f32_16x16x32_f16(false, a, false, b, (short)0, c,
                                                false, false);
}

// Load a 32x16 f16 B-fragment from an LDS weight stored row-major [n][RS]:
// B(k,n) = W[nb*16 + n][ks*32 + k].
// Lane layout: n = lane&15, K = (lane>>4)*8 + e (e<8), 16 + (lane>>4)*8 + (e-8) (e>=8).
__device__ __forceinline__ v16h bfrag(const _Float16* W, int RS, int nb, int ks,
                                      int lane) {
  int n  = nb * 16 + (lane & 15);
  int k0 = ks * 32 + (lane >> 4) * 8;
  const _Float16* p = W + n * RS + k0;
  v8h lo = *(const v8h*)(p);
  v8h hi = *(const v8h*)(p + 16);
  v16h r;
#pragma unroll
  for (int e = 0; e < 8; ++e) { r[e] = lo[e]; r[8 + e] = hi[e]; }
  return r;
}

__device__ __forceinline__ void cvt8(v16h& f, int base, float4 a, float4 b) {
  f[base + 0] = (_Float16)a.x; f[base + 1] = (_Float16)a.y;
  f[base + 2] = (_Float16)a.z; f[base + 3] = (_Float16)a.w;
  f[base + 4] = (_Float16)b.x; f[base + 5] = (_Float16)b.y;
  f[base + 6] = (_Float16)b.z; f[base + 7] = (_Float16)b.w;
}

// ---- Kernel 1: build the f16 weight blob once (single workgroup) ----
__global__ __launch_bounds__(256)
void ren_prep_kernel(const float* __restrict__ A,   const float* __restrict__ B1,
                     const float* __restrict__ B2,  const float* __restrict__ C1,
                     const float* __restrict__ D11, const float* __restrict__ D12,
                     const float* __restrict__ bv,  const float* __restrict__ bx,
                     unsigned char* __restrict__ ws)
{
  const int tid = threadIdx.x;
  _Float16* wC1   = (_Float16*)(ws + OFF_C1);
  _Float16* wA    = (_Float16*)(ws + OFF_A);
  _Float16* wB1   = (_Float16*)(ws + OFF_B1);
  _Float16* wD12  = (_Float16*)(ws + OFF_D12);
  _Float16* wB2   = (_Float16*)(ws + OFF_B2);
  _Float16* wD11m = (_Float16*)(ws + OFF_D11M);
  float*    wDdiag= (float*)(ws + OFF_DDIAG);
  float*    wBv   = (float*)(ws + OFF_BV);
  float*    wBx   = (float*)(ws + OFF_BX);

  for (int i = tid; i < 4096; i += 256) {
    wC1[i] = (_Float16)C1[i];
    wA [i] = (_Float16)A [i];
    wB1[i] = (_Float16)B1[i];
  }
  for (int i = tid; i < 64 * 32; i += 256) {
    int n = i >> 5, k = i & 31;
    wD12[i] = (k < 16) ? (_Float16)D12[n * 16 + k] : (_Float16)0.f;
    wB2 [i] = (k < 16) ? (_Float16)B2 [n * 16 + k] : (_Float16)0.f;
  }
  for (int i = tid; i < 4096; i += 256) {
    int b = i >> 10, r = i & 1023;
    int n = r >> 6, j = r & 63;
    wD11m[i] = (j < 16 * b) ? (_Float16)D11[(16 * b + n) * 64 + j]
                            : (_Float16)0.f;
  }
  for (int i = tid; i < 1024; i += 256) {
    int b = i >> 8, r = i & 255;
    int jj = r >> 4, ii = r & 15;
    wDdiag[i] = (ii > jj) ? D11[(16 * b + ii) * 64 + (16 * b + jj)] : 0.f;
  }
  if (tid < 64) { wBv[tid] = bv[tid]; wBx[tid] = bx[tid]; }
}

// ---- Kernel 2: main fused REN forward ----
__global__ __launch_bounds__(256)
void ren_fwd_kernel(const float* __restrict__ x, const float* __restrict__ u,
                    unsigned char* __restrict__ ws, float* __restrict__ out)
{
  __shared__ __align__(16) unsigned char sBlob[BLOB_BYTES];
  __shared__ __align__(16) float sTile[8][16 * 17];

  const int tid = threadIdx.x;

  // ---- Stage the pre-converted weight blob into LDS (async copy path) ----
#if defined(HAVE_ASYNC_LDS)
  for (int off = tid * 16; off < BLOB_BYTES; off += 256 * 16) {
    __builtin_amdgcn_global_load_async_to_lds_b128(
        (__attribute__((address_space(1))) v4i*)(ws + off),
        (__attribute__((address_space(3))) v4i*)(sBlob + off), 0, 0);
  }
#if __has_builtin(__builtin_amdgcn_s_wait_asynccnt)
  __builtin_amdgcn_s_wait_asynccnt(0);
#else
  asm volatile("s_wait_asynccnt 0" ::: "memory");
#endif
#else
  for (int off = tid * 16; off < BLOB_BYTES; off += 256 * 16) {
    *(float4*)(sBlob + off) = *(const float4*)(ws + off);
  }
#endif
  __syncthreads();

  const _Float16* sC1   = (const _Float16*)(sBlob + OFF_C1);
  const _Float16* sA    = (const _Float16*)(sBlob + OFF_A);
  const _Float16* sB1   = (const _Float16*)(sBlob + OFF_B1);
  const _Float16* sD12  = (const _Float16*)(sBlob + OFF_D12);
  const _Float16* sB2   = (const _Float16*)(sBlob + OFF_B2);
  const _Float16* sD11m = (const _Float16*)(sBlob + OFF_D11M);
  const float*    sDdiag= (const float*)(sBlob + OFF_DDIAG);
  const float*    sBv   = (const float*)(sBlob + OFF_BV);
  const float*    sBx   = (const float*)(sBlob + OFF_BX);

  const int lane = tid & 31;
  const int wv   = tid >> 5;
  const int kh   = lane >> 4;       // which K-half this lane holds (A layout)
  const int ln   = lane & 15;
  const int m0   = blockIdx.x * ROWS_PER_WG + wv * 16;
  const int mrow = m0 + ln;

  // ---- A-fragments for x (16x32 per k-step) and u (K padded to 32) ----
  // A layout: lane ln holds row M=ln; elems 0..7 -> K = ks*32 + kh*8 + e,
  //           elems 8..15 -> K = ks*32 + 16 + kh*8 + (e-8).
  v16h fx[2];
#pragma unroll
  for (int ks = 0; ks < 2; ++ks) {
    const float* px = x + (size_t)mrow * 64 + ks * 32 + kh * 8;
    float4 a0 = *(const float4*)(px);
    float4 a1 = *(const float4*)(px + 4);
    float4 a2 = *(const float4*)(px + 16);
    float4 a3 = *(const float4*)(px + 20);
    cvt8(fx[ks], 0, a0, a1);
    cvt8(fx[ks], 8, a2, a3);
  }
  v16h fu{};
  {
    const float* pu = u + (size_t)mrow * 16 + kh * 8;
    float4 u0 = *(const float4*)(pu);
    float4 u1 = *(const float4*)(pu + 4);
    cvt8(fu, 0, u0, u1);            // elems 8..15 (K=16..31) stay zero
  }

  // ---- Dense GEMMs: base = x@C1^T + u@D12^T + bv ; out = x@A^T + u@B2^T ----
  v8f base_acc[4], out_acc[4];
#pragma unroll
  for (int nb = 0; nb < 4; ++nb) {
    v8f c{};
    c = wmma_f16(fx[0], bfrag(sC1, 64, nb, 0, lane), c);
    c = wmma_f16(fx[1], bfrag(sC1, 64, nb, 1, lane), c);
    c = wmma_f16(fu,    bfrag(sD12, 32, nb, 0, lane), c);
    float bvn = sBv[nb * 16 + ln];
#pragma unroll
    for (int e = 0; e < 8; ++e) c[e] += bvn;
    base_acc[nb] = c;

    v8f o{};
    o = wmma_f16(fx[0], bfrag(sA, 64, nb, 0, lane), o);
    o = wmma_f16(fx[1], bfrag(sA, 64, nb, 1, lane), o);
    o = wmma_f16(fu,    bfrag(sB2, 32, nb, 0, lane), o);
    out_acc[nb] = o;
  }

  // ---- Blocked forward substitution for w ----
  // w kept purely in registers as growing A-fragments fw0 (cols 0..31) and
  // fw1 (cols 32..63); unfilled columns are zero so WMMA updates are exact.
  v16h fw0{}, fw1{};
  float* tile = sTile[wv];

#pragma unroll
  for (int b = 0; b < 4; ++b) {
    v8f acc = base_acc[b];
    if (b >= 1) acc = wmma_f16(fw0, bfrag(sD11m + b * 1024, 64, 0, 0, lane), acc);
    if (b == 3) acc = wmma_f16(fw1, bfrag(sD11m + b * 1024, 64, 0, 1, lane), acc);

    // Transpose 16x16 acc tile through LDS: C/D layout elem e is
    // (row m = e + 8*kh, col n = ln)  ->  tile[n][m]
    __syncthreads();
#pragma unroll
    for (int e = 0; e < 8; ++e) tile[ln * 17 + e + 8 * kh] = acc[e];
    __syncthreads();

    // Recursion mapping: lane owns row m'=ln, in-block cols i = kh*8 + t.
    float a[8];
#pragma unroll
    for (int t = 0; t < 8; ++t) a[t] = tile[(kh * 8 + t) * 17 + ln];

#pragma unroll
    for (int j = 0; j < 16; ++j) {
      const int oh = j >> 3, jl = j & 7;
      float wj = (kh == oh) ? fmaxf(a[jl], 0.f) : 0.f;
      wj += __shfl_xor(wj, 16, 32);         // broadcast within lane pair
      const float4 c0 = *(const float4*)(sDdiag + (b * 16 + j) * 16 + kh * 8);
      const float4 c1 = *(const float4*)(sDdiag + (b * 16 + j) * 16 + kh * 8 + 4);
      a[0] += wj * c0.x; a[1] += wj * c0.y; a[2] += wj * c0.z; a[3] += wj * c0.w;
      a[4] += wj * c1.x; a[5] += wj * c1.y; a[6] += wj * c1.z; a[7] += wj * c1.w;
    }

    // Fill w fragment: block b -> fw[b>>1], elems [0..8) if b even, [8..16) odd
    if ((b & 1) == 0) {
      v16h& f = (b == 0) ? fw0 : fw1;
#pragma unroll
      for (int t = 0; t < 8; ++t) f[t] = (_Float16)fmaxf(a[t], 0.f);
    } else {
      v16h& f = (b == 1) ? fw0 : fw1;
#pragma unroll
      for (int t = 0; t < 8; ++t) f[8 + t] = (_Float16)fmaxf(a[t], 0.f);
    }
  }

  // ---- out += w@B1^T + bx ; store ----
#pragma unroll
  for (int nb = 0; nb < 4; ++nb) {
    v8f o = out_acc[nb];
    o = wmma_f16(fw0, bfrag(sB1, 64, nb, 0, lane), o);
    o = wmma_f16(fw1, bfrag(sB1, 64, nb, 1, lane), o);
    float bxn = sBx[nb * 16 + ln];
    float* po = out + (size_t)(m0 + 8 * kh) * 64 + nb * 16 + ln;
#pragma unroll
    for (int e = 0; e < 8; ++e) po[(size_t)e * 64] = o[e] + bxn;
  }
}

extern "C" void kernel_launch(void* const* d_in, const int* in_sizes, int n_in,
                              void* d_out, int out_size, void* d_ws, size_t ws_size,
                              hipStream_t stream) {
  (void)in_sizes; (void)n_in; (void)ws_size; (void)out_size;
  const float* x   = (const float*)d_in[0];
  const float* u   = (const float*)d_in[1];
  const float* A   = (const float*)d_in[2];
  const float* B1  = (const float*)d_in[3];
  const float* B2  = (const float*)d_in[4];
  const float* C1  = (const float*)d_in[5];
  const float* D11 = (const float*)d_in[6];
  const float* D12 = (const float*)d_in[7];
  const float* bv  = (const float*)d_in[8];
  const float* bx  = (const float*)d_in[9];
  unsigned char* ws = (unsigned char*)d_ws;

  hipLaunchKernelGGL(ren_prep_kernel, dim3(1), dim3(256), 0, stream,
                     A, B1, B2, C1, D11, D12, bv, bx, ws);
  hipLaunchKernelGGL(ren_fwd_kernel, dim3(NTOT / ROWS_PER_WG), dim3(256), 0,
                     stream, x, u, ws, (float*)d_out);
}
